// PatchcoreModel_51814485458959
// MI455X (gfx1250) — compile-verified, matrix operand order checked by reading, not run on hardware
//
#include <hip/hip_runtime.h>
#include <math.h>

// ---------------- problem constants ----------------
#define BB   16
#define DD   768
#define HH   28
#define WW   28
#define NN   (BB * HH * WW)      // 12544 query rows
#define MM   16384               // memory bank rows
#define KB   (DD / 32)           // 24 k-blocks of 32
#define NT   (NN / 16)           // 784 row tiles
#define MT   (MM / 16)           // 1024 col tiles
#define COLSPLIT 4
#define PANELS_PER_SPLIT (MT / 8 / COLSPLIT)   // 32 panels (8 col-tiles each)
#define AMAP (BB * 224 * 224)    // 802816
#define F_INF 3.402823466e38f

typedef unsigned short u16;
typedef __attribute__((ext_vector_type(16))) __bf16 v16bf;
typedef __attribute__((ext_vector_type(8)))  float  v8f;
typedef __attribute__((ext_vector_type(4)))  unsigned int u32x4;
typedef __attribute__((ext_vector_type(8)))  int i32x8;
typedef __attribute__((ext_vector_type(4)))  int i32x4;

#if defined(__HIP_DEVICE_COMPILE__) && \
    __has_builtin(__builtin_amdgcn_tensor_load_to_lds) && \
    __has_builtin(__builtin_amdgcn_s_wait_tensorcnt)
#define USE_TDM 1
#else
#define USE_TDM 0
#endif

// ---------------- helpers ----------------
__device__ inline u16 f2bf(float x) {           // round-to-nearest-even truncation
    unsigned u = __float_as_uint(x);
    u += 0x7FFFu + ((u >> 16) & 1u);
    return (u16)(u >> 16);
}
__device__ inline float bf2f(u16 h) { return __uint_as_float(((unsigned)h) << 16); }

// fragment lane mapping for 16-bit 16x32 A / 32x16 B tiles (wave32):
//   kk = 16*c + 8*h + jj  ->  lane = m + 16*h, slot j = 8*c + jj
__device__ inline void frag_pos(int m, int kk, int& lane, int& j) {
    int c = (kk >> 4) & 1, h = (kk >> 3) & 1, jj = kk & 7;
    lane = m + 16 * h;
    j = 8 * c + jj;
}

// ---------------- kernel 1: avg-pool 3x3 (zero pad, /9) + pack A hi/lo ----------------
__global__ __launch_bounds__(256) void pool_pack_a_kernel(
    const float* __restrict__ feat, float* __restrict__ emb, u16* __restrict__ apk)
{
    long long i = (long long)blockIdx.x * 256 + threadIdx.x;
    if (i >= (long long)NN * DD) return;
    int d = (int)(i % DD);
    int n = (int)(i / DD);
    int b = n / (HH * WW);
    int rem = n % (HH * WW);
    int y = rem / WW, x = rem % WW;

    const float* base = feat + ((size_t)(b * DD + d)) * (HH * WW);
    float s = 0.0f;
    #pragma unroll
    for (int dy = -1; dy <= 1; ++dy) {
        int yy = y + dy;
        if (yy < 0 || yy >= HH) continue;
        #pragma unroll
        for (int dx = -1; dx <= 1; ++dx) {
            int xx = x + dx;
            if (xx < 0 || xx >= WW) continue;
            s += base[yy * WW + xx];
        }
    }
    float p = s * (1.0f / 9.0f);
    emb[(size_t)n * DD + d] = p;

    int rowTile = n >> 4, m = n & 15;
    int kblk = d >> 5, kk = d & 31;
    int lane, j; frag_pos(m, kk, lane, j);
    u16 hi = f2bf(p);
    u16 lo = f2bf(p - bf2f(hi));
    size_t tb = (size_t)(rowTile * KB + kblk) * 1024;   // [rowTile][kblk][hi 512 | lo 512]
    apk[tb + (size_t)lane * 16 + j]       = hi;
    apk[tb + 512 + (size_t)lane * 16 + j] = lo;
}

// ---------------- kernel 2: pack B hi/lo, layout [kblk][colTile][hi|lo] ----------------
__global__ __launch_bounds__(256) void pack_b_kernel(
    const float* __restrict__ mb, u16* __restrict__ bpk)
{
    long long i = (long long)blockIdx.x * 256 + threadIdx.x;
    if (i >= (long long)MM * DD) return;
    int d = (int)(i % DD);
    int r = (int)(i / DD);
    float p = mb[i];
    int colTile = r >> 4, m = r & 15;
    int kblk = d >> 5, kk = d & 31;
    int lane, j; frag_pos(m, kk, lane, j);
    u16 hi = f2bf(p);
    u16 lo = f2bf(p - bf2f(hi));
    size_t tb = (size_t)(kblk * MT + colTile) * 1024;
    bpk[tb + (size_t)lane * 16 + j]       = hi;
    bpk[tb + 512 + (size_t)lane * 16 + j] = lo;
}

// ---------------- kernel 3: row sum of squares ----------------
__global__ __launch_bounds__(256) void rowsq_kernel(
    const float* __restrict__ src, float* __restrict__ out, int cols)
{
    __shared__ float red[256];
    int row = blockIdx.x, tid = threadIdx.x;
    const float* p = src + (size_t)row * cols;
    float s = 0.0f;
    for (int k = tid; k < cols; k += 256) { float v = p[k]; s += v * v; }
    red[tid] = s; __syncthreads();
    for (int st = 128; st > 0; st >>= 1) {
        if (tid < st) red[tid] += red[tid + st];
        __syncthreads();
    }
    if (tid == 0) out[row] = red[0];
}

// ---------------- kernel 4: init packed (min-dist2, argmin) combine buffer ----------------
__global__ __launch_bounds__(256) void init_packed_kernel(unsigned long long* p, int n)
{
    int i = blockIdx.x * 256 + threadIdx.x;
    if (i < n) p[i] = 0xFFFFFFFFFFFFFFFFull;
}

#if USE_TDM
// TDM: DMA one contiguous 16KB chunk (4096 x 4B) from global into LDS.
__device__ inline void tdm_issue(const u16* gsrc, u16* lds_dst)
{
    unsigned long long ga = (unsigned long long)(size_t)gsrc;
    unsigned lds = (unsigned)(size_t)lds_dst;         // low 32 bits = LDS byte offset
    u32x4 g0;
    g0[0] = 1u;                                       // count=1 valid descriptor
    g0[1] = lds;                                      // lds_addr
    g0[2] = (unsigned)ga;                             // global_addr[31:0]
    g0[3] = (unsigned)((ga >> 32) & 0x1FFFFFFu) | 0x80000000u; // ga[56:32] | type=2
    i32x8 g1;
    g1[0] = (2 << 16);                                // data_size = 4B
    g1[1] = (int)(4096u << 16);                       // tensor_dim0[15:0] = 4096
    g1[2] = (1 << 16);                                // tensor_dim0 hi=0, tensor_dim1=1
    g1[3] = (int)(4096u << 16);                       // tile_dim0 = 4096
    g1[4] = 0;                                        // tile_dim1/2 unused
    g1[5] = 4096;                                     // tensor_dim0_stride
    g1[6] = 0; g1[7] = 0;
    i32x4 z4 = {0, 0, 0, 0};
    i32x8 z8 = {0, 0, 0, 0, 0, 0, 0, 0};
    // 6-arg toolchain variant: (g0, g1, g2, g3, g4, cpol)
    __builtin_amdgcn_tensor_load_to_lds(g0, g1, z4, z4, z8, 0);
}
#endif

// ---------------- kernel 5: fused GEMM(bf16x3 WMMA) + row min/argmin ----------------
// grid (NT/8, COLSPLIT) x 256 threads (8 waves); wave w handles row tile blockIdx.x*8+w.
__global__ __launch_bounds__(256) void dist_min_kernel(
    const u16* __restrict__ apk, const u16* __restrict__ bpk,
    const float* __restrict__ a2, const float* __restrict__ b2,
    unsigned long long* __restrict__ packed)
{
    __shared__ __align__(32) u16 smem[2][8192];   // 2 x 16KB B-panel chunks
    const int tid = threadIdx.x;
    const int lane = tid & 31;
    const int wave = tid >> 5;
    const int rowTile = blockIdx.x * 8 + wave;
    const int panel0 = blockIdx.y * PANELS_PER_SPLIT;
    const int TOTAL = PANELS_PER_SPLIT * KB;
    const int h = lane >> 4;                      // C/D half: rows v (h=0) / v+8 (h=1)

    float a2v[8];
    #pragma unroll
    for (int v = 0; v < 8; ++v) a2v[v] = a2[rowTile * 16 + v + 8 * h];

    float runMin[8];
    int   runIdx[8];
    #pragma unroll
    for (int v = 0; v < 8; ++v) { runMin[v] = F_INF; runIdx[v] = 0; }

    // chunk(it): panel p = panel0 + it/KB, kblk = it%KB; 8 col-tiles, hi+lo -> 16KB
    #define CHUNK_SRC(it) (bpk + (size_t)(((it) % KB) * MT + ((panel0 + (it) / KB) * 8)) * 1024)

#if USE_TDM
    if (wave == 0) tdm_issue(CHUNK_SRC(0), &smem[0][0]);
#endif

    int it = 0;
    for (int pl = 0; pl < PANELS_PER_SPLIT; ++pl) {
        const int panel = panel0 + pl;
        v8f zero = {0.f, 0.f, 0.f, 0.f, 0.f, 0.f, 0.f, 0.f};
        v8f acc[8];
        #pragma unroll
        for (int c = 0; c < 8; ++c) acc[c] = zero;

        for (int k = 0; k < KB; ++k, ++it) {
#if USE_TDM
            if (wave == 0) {
                if (it + 1 < TOTAL) {
                    tdm_issue(CHUNK_SRC(it + 1), &smem[(it + 1) & 1][0]);
                    __builtin_amdgcn_s_wait_tensorcnt(1);   // chunk `it` complete (in-order)
                } else {
                    __builtin_amdgcn_s_wait_tensorcnt(0);
                }
            }
            __syncthreads();                                // publish LDS chunk to all waves
#else
            {   // cooperative fallback copy
                const uint4* gs = (const uint4*)CHUNK_SRC(it);
                uint4* dch = (uint4*)&smem[it & 1][0];
                for (int jj = tid; jj < 1024; jj += 256) dch[jj] = gs[jj];
            }
            __syncthreads();
#endif
            const u16* sb = &smem[it & 1][0];
            const u16* ab = apk + (size_t)(rowTile * KB + k) * 1024;
            v16bf ah = reinterpret_cast<const v16bf*>(ab)[lane];
            v16bf al = reinterpret_cast<const v16bf*>(ab + 512)[lane];

            #pragma unroll
            for (int c = 0; c < 8; ++c) {
                v16bf bh = reinterpret_cast<const v16bf*>(sb + c * 1024)[lane];
                v16bf bl = reinterpret_cast<const v16bf*>(sb + c * 1024 + 512)[lane];
                // bf16x3 compensated product: hi*hi + hi*lo + lo*hi
                acc[c] = __builtin_amdgcn_wmma_f32_16x16x32_bf16(
                            false, ah, false, bh, (short)0, acc[c], false, false);
                acc[c] = __builtin_amdgcn_wmma_f32_16x16x32_bf16(
                            false, ah, false, bl, (short)0, acc[c], false, false);
                acc[c] = __builtin_amdgcn_wmma_f32_16x16x32_bf16(
                            false, al, false, bh, (short)0, acc[c], false, false);
            }
            __syncthreads();   // all waves done with this buffer before TDM overwrites it
        }

        // panel epilogue: d2 = ||a||^2 + ||b||^2 - 2 a.b ; fold into running min
        #pragma unroll
        for (int c = 0; c < 8; ++c) {
            int col = panel * 128 + c * 16 + (lane & 15);
            float bb = b2[col];
            #pragma unroll
            for (int v = 0; v < 8; ++v) {
                float d2 = a2v[v] + bb - 2.0f * acc[c][v];
                if (d2 < runMin[v]) { runMin[v] = d2; runIdx[v] = col; }
            }
        }
    }
    #undef CHUNK_SRC

    // combine across lanes / col-splits: lexicographic (dist2 bits, col) atomicMin.
    #pragma unroll
    for (int v = 0; v < 8; ++v) {
        int row = rowTile * 16 + v + 8 * h;
        float d2c = fmaxf(runMin[v], 0.0f);      // dist2 >= 0 -> bit order == float order
        unsigned long long pk =
            (((unsigned long long)__float_as_uint(d2c)) << 32) | (unsigned)runIdx[v];
        atomicMin(&packed[row], pk);
    }
}

// ---------------- kernel 6: finalize min -> patch_scores / locations ----------------
__global__ __launch_bounds__(256) void finalize_min_kernel(
    const unsigned long long* __restrict__ packed,
    float* __restrict__ ps, int* __restrict__ loc, int n)
{
    int i = blockIdx.x * 256 + threadIdx.x;
    if (i >= n) return;
    unsigned long long pk = packed[i];
    ps[i]  = sqrtf(__uint_as_float((unsigned)(pk >> 32)));
    loc[i] = (int)(pk & 0xFFFFFFFFu);
}

// ---------------- kernel 7: per-batch argmax (first-index ties) ----------------
__global__ __launch_bounds__(256) void batch_argmax_kernel(
    const float* __restrict__ ps, const int* __restrict__ loc,
    float* __restrict__ score, int* __restrict__ maxIdx, int* __restrict__ nnIdx)
{
    __shared__ float sv[256];
    __shared__ int   si[256];
    int b = blockIdx.x, tid = threadIdx.x;
    float best = -F_INF; int bi = 0;
    for (int i = tid; i < HH * WW; i += 256) {
        float v = ps[b * (HH * WW) + i];
        if (v > best) { best = v; bi = i; }
    }
    sv[tid] = best; si[tid] = bi; __syncthreads();
    for (int s = 128; s > 0; s >>= 1) {
        if (tid < s) {
            if (sv[tid + s] > sv[tid] ||
                (sv[tid + s] == sv[tid] && si[tid + s] < si[tid])) {
                sv[tid] = sv[tid + s]; si[tid] = si[tid + s];
            }
        }
        __syncthreads();
    }
    if (tid == 0) {
        score[b]  = sv[0];
        maxIdx[b] = si[0];
        nnIdx[b]  = loc[b * (HH * WW) + si[0]];
    }
}

// ---------------- kernel 8: d2(nn_sample, memory_bank) ----------------
__global__ __launch_bounds__(256) void nn_dist_kernel(
    const float* __restrict__ mb, const int* __restrict__ nnIdx, float* __restrict__ d2b)
{
    __shared__ float nn[DD];
    int b = blockIdx.y;
    int col = blockIdx.x * 256 + threadIdx.x;
    const float* src = mb + (size_t)nnIdx[b] * DD;
    for (int i = threadIdx.x; i < DD; i += 256) nn[i] = src[i];
    __syncthreads();
    const float* m = mb + (size_t)col * DD;
    float s = 0.0f;
    for (int k = 0; k < DD; ++k) { float df = nn[k] - m[k]; s += df * df; }
    d2b[(size_t)b * MM + col] = s;     // squared dist: same ordering as sqrt'd
}

// ---------------- kernel 9: top-9 smallest (selection) ----------------
__global__ __launch_bounds__(256) void topk_kernel(
    float* __restrict__ d2b, int* __restrict__ support)
{
    __shared__ float sv[256];
    __shared__ int   si[256];
    int b = blockIdx.x, tid = threadIdx.x;
    float* row = d2b + (size_t)b * MM;
    for (int r = 0; r < 9; ++r) {
        float best = F_INF; int bi = 0;
        for (int i = tid; i < MM; i += 256) {
            float v = row[i];
            if (v < best) { best = v; bi = i; }
        }
        sv[tid] = best; si[tid] = bi; __syncthreads();
        for (int s = 128; s > 0; s >>= 1) {
            if (tid < s) {
                if (sv[tid + s] < sv[tid] ||
                    (sv[tid + s] == sv[tid] && si[tid + s] < si[tid])) {
                    sv[tid] = sv[tid + s]; si[tid] = si[tid + s];
                }
            }
            __syncthreads();
        }
        if (tid == 0) { support[b * 9 + r] = si[0]; row[si[0]] = F_INF; }
        __syncthreads();
    }
}

// ---------------- kernel 10: softmax re-weighting -> anomaly score ----------------
__global__ __launch_bounds__(256) void final_score_kernel(
    const float* __restrict__ emb, const float* __restrict__ mb,
    const int* __restrict__ support, const int* __restrict__ maxIdx,
    const float* __restrict__ score, float* __restrict__ out_scores)
{
    __shared__ float red[256];
    __shared__ float dists[9];
    int b = blockIdx.x, tid = threadIdx.x;
    const float* mf = emb + ((size_t)b * (HH * WW) + maxIdx[b]) * DD;
    for (int r = 0; r < 9; ++r) {
        const float* sp = mb + (size_t)support[b * 9 + r] * DD;
        float s = 0.0f;
        for (int k = tid; k < DD; k += 256) { float df = mf[k] - sp[k]; s += df * df; }
        red[tid] = s; __syncthreads();
        for (int st = 128; st > 0; st >>= 1) {
            if (tid < st) red[tid] += red[tid + st];
            __syncthreads();
        }
        if (tid == 0) dists[r] = sqrtf(fmaxf(red[0], 0.0f));
        __syncthreads();
    }
    if (tid == 0) {
        float m = dists[0];
        for (int r = 1; r < 9; ++r) m = fmaxf(m, dists[r]);
        float sum = 0.0f, e0 = 0.0f;
        for (int r = 0; r < 9; ++r) {
            float e = expf(dists[r] - m);
            sum += e;
            if (r == 0) e0 = e;
        }
        out_scores[b] = (1.0f - e0 / sum) * score[b];
    }
}

// ---------------- kernel 11: bilinear 28x28 -> 224x224 (half-pixel, clamp) ----------------
__global__ __launch_bounds__(256) void resize_kernel(
    const float* __restrict__ ps, float* __restrict__ rs)
{
    int o = blockIdx.x * 256 + threadIdx.x;
    if (o >= AMAP) return;
    int b = o / (224 * 224);
    int rem = o % (224 * 224);
    int y = rem / 224, x = rem % 224;
    float sy = (y + 0.5f) * 0.125f - 0.5f;
    float sx = (x + 0.5f) * 0.125f - 0.5f;
    int y0 = (int)floorf(sy); float fy = sy - (float)y0;
    int x0 = (int)floorf(sx); float fx = sx - (float)x0;
    int y1 = y0 + 1, x1 = x0 + 1;
    y0 = y0 < 0 ? 0 : (y0 > 27 ? 27 : y0);
    y1 = y1 < 0 ? 0 : (y1 > 27 ? 27 : y1);
    x0 = x0 < 0 ? 0 : (x0 > 27 ? 27 : x0);
    x1 = x1 < 0 ? 0 : (x1 > 27 ? 27 : x1);
    const float* p = ps + b * (HH * WW);
    float v = (1.0f - fy) * ((1.0f - fx) * p[y0 * 28 + x0] + fx * p[y0 * 28 + x1]) +
              fy *          ((1.0f - fx) * p[y1 * 28 + x0] + fx * p[y1 * 28 + x1]);
    rs[o] = v;
}

// ---------------- kernels 12/13: separable 33-tap Gaussian (sigma=4, reflect pad) ----------------
__device__ inline void gauss_weights(float* g, int tid) {
    if (tid == 0) {
        float sum = 0.0f;
        for (int k = 0; k < 33; ++k) {
            float c = (float)(k - 16);
            float e = expf(-(c * c) / 32.0f);   // 2*sigma^2 = 32
            g[k] = e; sum += e;
        }
        for (int k = 0; k < 33; ++k) g[k] /= sum;
    }
}
__device__ inline int reflect224(int i) { return i < 0 ? -i : (i > 223 ? 446 - i : i); }

__global__ __launch_bounds__(256) void blur_h_kernel(
    const float* __restrict__ in, float* __restrict__ out)
{
    __shared__ float g[33];
    gauss_weights(g, threadIdx.x);
    __syncthreads();
    int o = blockIdx.x * 256 + threadIdx.x;
    if (o >= AMAP) return;
    int b = o / (224 * 224);
    int rem = o % (224 * 224);
    int y = rem / 224, x = rem % 224;
    const float* row = in + (size_t)b * 224 * 224 + y * 224;
    float s = 0.0f;
    for (int k = 0; k < 33; ++k) s += g[k] * row[reflect224(x + k - 16)];
    out[o] = s;
}

__global__ __launch_bounds__(256) void blur_v_kernel(
    const float* __restrict__ in, float* __restrict__ out)
{
    __shared__ float g[33];
    gauss_weights(g, threadIdx.x);
    __syncthreads();
    int o = blockIdx.x * 256 + threadIdx.x;
    if (o >= AMAP) return;
    int b = o / (224 * 224);
    int rem = o % (224 * 224);
    int y = rem / 224, x = rem % 224;
    const float* img = in + (size_t)b * 224 * 224;
    float s = 0.0f;
    for (int k = 0; k < 33; ++k) s += g[k] * img[reflect224(y + k - 16) * 224 + x];
    out[o] = s;
}

// ---------------- host launcher ----------------
extern "C" void kernel_launch(void* const* d_in, const int* in_sizes, int n_in,
                              void* d_out, int out_size, void* d_ws, size_t ws_size,
                              hipStream_t stream)
{
    const float* feat = (const float*)d_in[0];   // [16,768,28,28]
    const float* mb   = (const float*)d_in[1];   // [16384,768]
    float* out = (float*)d_out;                  // 802816 amap + 16 scores

    char* w = (char*)d_ws;
    auto alloc = [&](size_t bytes) -> void* {
        void* p = (void*)w;
        w += (bytes + 255) & ~(size_t)255;
        return p;
    };
    float* emb = (float*)alloc((size_t)NN * DD * 4);               // 38.5 MB
    u16*   apk = (u16*)alloc((size_t)NN * DD * 2 * 2);             // 38.5 MB
    u16*   bpk = (u16*)alloc((size_t)MM * DD * 2 * 2);             // 50.3 MB
    float* a2  = (float*)alloc((size_t)NN * 4);
    float* b2  = (float*)alloc((size_t)MM * 4);
    unsigned long long* packed = (unsigned long long*)alloc((size_t)NN * 8);
    float* ps  = (float*)alloc((size_t)NN * 4);
    int*   loc = (int*)alloc((size_t)NN * 4);
    float* score  = (float*)alloc(16 * 4);
    int*   maxIdx = (int*)alloc(16 * 4);
    int*   nnIdx  = (int*)alloc(16 * 4);
    float* d2b = (float*)alloc((size_t)16 * MM * 4);               // 1 MB
    int*   support = (int*)alloc(16 * 9 * 4);
    float* rs  = (float*)alloc((size_t)AMAP * 4);                  // 3.2 MB
    float* tb  = (float*)alloc((size_t)AMAP * 4);                  // 3.2 MB

    pool_pack_a_kernel<<<(NN * DD) / 256, 256, 0, stream>>>(feat, emb, apk);
    pack_b_kernel<<<(MM * DD) / 256, 256, 0, stream>>>(mb, bpk);
    rowsq_kernel<<<NN, 256, 0, stream>>>(emb, a2, DD);
    rowsq_kernel<<<MM, 256, 0, stream>>>(mb, b2, DD);
    init_packed_kernel<<<(NN + 255) / 256, 256, 0, stream>>>(packed, NN);

    dist_min_kernel<<<dim3(NT / 8, COLSPLIT), 256, 0, stream>>>(apk, bpk, a2, b2, packed);

    finalize_min_kernel<<<(NN + 255) / 256, 256, 0, stream>>>(packed, ps, loc, NN);
    batch_argmax_kernel<<<16, 256, 0, stream>>>(ps, loc, score, maxIdx, nnIdx);
    nn_dist_kernel<<<dim3(MM / 256, 16), 256, 0, stream>>>(mb, nnIdx, d2b);
    topk_kernel<<<16, 256, 0, stream>>>(d2b, support);
    final_score_kernel<<<16, 256, 0, stream>>>(emb, mb, support, maxIdx, score, out + AMAP);

    resize_kernel<<<AMAP / 256, 256, 0, stream>>>(ps, rs);
    blur_h_kernel<<<AMAP / 256, 256, 0, stream>>>(rs, tb);
    blur_v_kernel<<<AMAP / 256, 256, 0, stream>>>(tb, out);
}
